// T5Attention_61194694033955
// MI455X (gfx1250) — matrix operand order repaired and checked by reading, not compiled, not run on hardware
//
#include <hip/hip_runtime.h>
#include <hip/hip_bf16.h>

// ---------------- problem constants ----------------
#define BB   2
#define SS   2048
#define DD   1024
#define HH   16
#define DK   64
#define NBUK 32
// H*DK == DD

typedef __attribute__((ext_vector_type(16))) __bf16 v16bf;
typedef __attribute__((ext_vector_type(8)))  float  v8f;

union FragBF {
    v16bf v;
    unsigned int u[8];
};

__device__ __forceinline__ unsigned short f32_to_bf16(float f) {
    unsigned int u = __float_as_uint(f);
    unsigned int r = u + 0x7FFFu + ((u >> 16) & 1u);
    return (unsigned short)(r >> 16);
}

// A-fragment (16x32 bf16) per-VGPR k offset, contiguous order k = 16*half + 2*v (matches B)
__device__ __forceinline__ int koffA(int v, int half) {
    return (half ? 16 : 0) + v * 2;
}

__device__ __forceinline__ v8f wmma_bf16(const FragBF& a, const FragBF& b, v8f c) {
    return __builtin_amdgcn_wmma_f32_16x16x32_bf16(
        /*neg_a=*/false, a.v, /*neg_b=*/false, b.v,
        /*c_mod=*/(short)0, c, /*reuse_a=*/false, /*reuse_b=*/false);
}

// ---------------- 1) f32 -> bf16 convert ----------------
__global__ void conv_bf16_kernel(const float* __restrict__ in,
                                 unsigned short* __restrict__ out, int n) {
    int i = blockIdx.x * blockDim.x + threadIdx.x;
    if (i < n) out[i] = f32_to_bf16(in[i]);
}

// ---------------- 2) W[K][N] f32 -> Wt[N][K] bf16 ----------------
__global__ void transpose_bf16_kernel(const float* __restrict__ W,
                                      unsigned short* __restrict__ Wt) {
    int i = blockIdx.x * blockDim.x + threadIdx.x;   // over 1024*1024
    int n = i >> 10;
    int k = i & 1023;
    Wt[(size_t)n * DD + k] = f32_to_bf16(W[(size_t)k * DD + n]);
}

// ---------------- 3) position_bias (exact T5 bucketing) ----------------
__global__ void bias_kernel(const float* __restrict__ table,
                            float* __restrict__ bias_out) {
    int i = blockIdx.x * blockDim.x + threadIdx.x;   // over S*S
    int q = i >> 11;           // / S
    int k = i & (SS - 1);      // % S
    int rp = k - q;                                  // mem - ctx
    int bucket = (rp > 0) ? 16 : 0;
    int ra = (rp < 0) ? -rp : rp;
    if (ra < 8) {
        bucket += ra;
    } else {
        // 8 + trunc(log(ra/8)/log(16) * 8), clamped to 15
        float t = __logf((float)ra * 0.125f) * (8.0f / 2.7725887222397811f);
        int rl = 8 + (int)t;
        bucket += (rl < 15) ? rl : 15;
    }
#pragma unroll
    for (int h = 0; h < HH; ++h) {
        bias_out[((size_t)h * SS + q) * SS + k] = table[bucket * HH + h];
    }
}

// ---------------- 4) fused Q/K/V projection GEMM (bf16 WMMA) ----------------
// Xh: [B*S, D] bf16 row-major, read ONCE for all three projections.
// WqT/WkT/WvT: [N, K] bf16 (transposed weights).
// Q,K out: [B,H,S,DK] bf16.  V out: [B,H,DK,S] bf16 (transposed for PV).
__global__ __launch_bounds__(256)
void fused_qkv_kernel(const unsigned short* __restrict__ Xh,
                      const unsigned short* __restrict__ WqT,
                      const unsigned short* __restrict__ WkT,
                      const unsigned short* __restrict__ WvT,
                      unsigned short* __restrict__ Qo,
                      unsigned short* __restrict__ Ko,
                      unsigned short* __restrict__ Vo) {
    int lane = threadIdx.x & 31;
    int wave = threadIdx.x >> 5;
    int half = (lane >> 4) & 1;
    int l15  = lane & 15;
    int mtile = blockIdx.x;                 // 0..255 over M=B*S
    int ntile = blockIdx.y * 8 + wave;      // 0..63 over N=1024

    const unsigned short* arow = Xh + (size_t)(mtile * 16 + l15) * DD;
    size_t brow_off = (size_t)(ntile * 16 + l15) * DD + (half ? 16 : 0);
    const unsigned short* bq = WqT + brow_off;
    const unsigned short* bk = WkT + brow_off;
    const unsigned short* bv = WvT + brow_off;

    v8f accq = {}, acck = {}, accv = {};
    for (int kk = 0; kk < DD; kk += 32) {
        __builtin_prefetch((const void*)(arow + kk + 256), 0, 1);
        FragBF a, b;
#pragma unroll
        for (int v = 0; v < 8; ++v)
            a.u[v] = *(const unsigned int*)(arow + kk + koffA(v, half));
#pragma unroll
        for (int v = 0; v < 8; ++v)
            b.u[v] = *(const unsigned int*)(bq + kk + v * 2);
        accq = wmma_bf16(a, b, accq);
#pragma unroll
        for (int v = 0; v < 8; ++v)
            b.u[v] = *(const unsigned int*)(bk + kk + v * 2);
        acck = wmma_bf16(a, b, acck);
#pragma unroll
        for (int v = 0; v < 8; ++v)
            b.u[v] = *(const unsigned int*)(bv + kk + v * 2);
        accv = wmma_bf16(a, b, accv);
    }

    int n  = ntile * 16 + l15;
    int h  = n >> 6;
    int dk = n & 63;
#pragma unroll
    for (int v = 0; v < 8; ++v) {
        int r = mtile * 16 + v + (half ? 8 : 0);
        int b_ = r >> 11;          // / S
        int s_ = r & (SS - 1);     // % S
        size_t off_qk = ((size_t)(b_ * HH + h) * SS + s_) * DK + dk;
        size_t off_v  = ((size_t)(b_ * HH + h) * DK + dk) * SS + s_;
        Qo[off_qk] = f32_to_bf16(accq[v]);
        Ko[off_qk] = f32_to_bf16(acck[v]);
        Vo[off_v]  = f32_to_bf16(accv[v]);
    }
}

// ---------------- 5) flash attention (bf16 WMMA, online softmax) ----------------
// Q,K: [B,H,S,DK] bf16.  Vt: [B,H,DK,S] bf16.  bias: [H,S,S] f32.
// ctx out: [B,S,H*DK] bf16.
__global__ __launch_bounds__(128)
void attention_kernel(const unsigned short* __restrict__ Q,
                      const unsigned short* __restrict__ Kh,
                      const unsigned short* __restrict__ Vt,
                      const float* __restrict__ bias,
                      unsigned short* __restrict__ ctx) {
    __shared__ unsigned short pstage[4][16 * 32];   // per-wave prob tile

    int lane = threadIdx.x & 31;
    int wave = threadIdx.x >> 5;
    int half = (lane >> 4) & 1;
    int l15  = lane & 15;

    int b = blockIdx.z;
    int h = blockIdx.y;
    int qbase = (blockIdx.x * 4 + wave) * 16;

    const unsigned short* Qb = Q  + (size_t)(b * HH + h) * SS * DK;
    const unsigned short* Kb = Kh + (size_t)(b * HH + h) * SS * DK;
    const unsigned short* Vb = Vt + (size_t)(b * HH + h) * DK * SS;
    const float* biasH = bias + (size_t)h * SS * SS;

    // Q A-fragments for the two 32-wide d chunks (reused across all 2048 keys)
    FragBF aq0, aq1;
    {
        const unsigned short* qrow = Qb + (size_t)(qbase + l15) * DK;
#pragma unroll
        for (int v = 0; v < 8; ++v) {
            aq0.u[v] = *(const unsigned int*)(qrow + koffA(v, half));
            aq1.u[v] = *(const unsigned int*)(qrow + 32 + koffA(v, half));
        }
    }

    float m_i[8], l_i[8];
#pragma unroll
    for (int v = 0; v < 8; ++v) { m_i[v] = -1e30f; l_i[v] = 0.0f; }
    v8f acc[4] = {{}, {}, {}, {}};

    for (int kbb = 0; kbb < SS; kbb += 32) {
        // ---- scores: two 16x16 tiles over 32 keys ----
        v8f s0 = {}, s1 = {};
        const unsigned short* k0p = Kb + (size_t)(kbb + l15) * DK + (half ? 16 : 0);
        const unsigned short* k1p = k0p + (size_t)16 * DK;
        __builtin_prefetch((const void*)(k0p + 32 * DK), 0, 1);
        FragBF bk;
#pragma unroll
        for (int v = 0; v < 8; ++v) bk.u[v] = *(const unsigned int*)(k0p + v * 2);
        s0 = wmma_bf16(aq0, bk, s0);
#pragma unroll
        for (int v = 0; v < 8; ++v) bk.u[v] = *(const unsigned int*)(k0p + 32 + v * 2);
        s0 = wmma_bf16(aq1, bk, s0);
#pragma unroll
        for (int v = 0; v < 8; ++v) bk.u[v] = *(const unsigned int*)(k1p + v * 2);
        s1 = wmma_bf16(aq0, bk, s1);
#pragma unroll
        for (int v = 0; v < 8; ++v) bk.u[v] = *(const unsigned int*)(k1p + 32 + v * 2);
        s1 = wmma_bf16(aq1, bk, s1);

        // ---- add relative-position bias (no 1/sqrt(dk) scaling in T5) ----
#pragma unroll
        for (int v = 0; v < 8; ++v) {
            int q = qbase + v + (half ? 8 : 0);
            s0[v] += biasH[(size_t)q * SS + kbb + l15];
            s1[v] += biasH[(size_t)q * SS + kbb + 16 + l15];
        }

        // ---- online softmax update (row stats replicated across 16-lane half) ----
        float alpha[8];
#pragma unroll
        for (int v = 0; v < 8; ++v) {
            float mx = fmaxf(s0[v], s1[v]);
#pragma unroll
            for (int off = 1; off < 16; off <<= 1)
                mx = fmaxf(mx, __shfl_xor(mx, off));
            float mn = fmaxf(m_i[v], mx);
            alpha[v] = __expf(m_i[v] - mn);
            float p0 = __expf(s0[v] - mn);
            float p1 = __expf(s1[v] - mn);
            s0[v] = p0; s1[v] = p1;
            float rs = p0 + p1;
#pragma unroll
            for (int off = 1; off < 16; off <<= 1)
                rs += __shfl_xor(rs, off);
            l_i[v] = l_i[v] * alpha[v] + rs;
            m_i[v] = mn;
        }
#pragma unroll
        for (int dt = 0; dt < 4; ++dt)
#pragma unroll
            for (int v = 0; v < 8; ++v) acc[dt][v] *= alpha[v];

        // ---- transpose prob tile C-layout -> A-layout through LDS ----
        unsigned short* pl = pstage[wave];
#pragma unroll
        for (int v = 0; v < 8; ++v) {
            int r = v + (half ? 8 : 0);
            pl[r * 32 + l15]      = f32_to_bf16(s0[v]);
            pl[r * 32 + 16 + l15] = f32_to_bf16(s1[v]);
        }
        asm volatile("s_wait_dscnt 0" ::: "memory");  // wave-private RAW fence
        FragBF pa;
        const unsigned int* pl32 = (const unsigned int*)pl;
#pragma unroll
        for (int v = 0; v < 8; ++v)
            pa.u[v] = pl32[l15 * 16 + (koffA(v, half) >> 1)];

        // ---- acc += P @ V  (4 d-tiles of 16, A-fragment reused 4x) ----
#pragma unroll
        for (int dt = 0; dt < 4; ++dt) {
            FragBF bv;
            const unsigned short* vp =
                Vb + (size_t)(dt * 16 + l15) * SS + kbb + (half ? 16 : 0);
#pragma unroll
            for (int v = 0; v < 8; ++v)
                bv.u[v] = *(const unsigned int*)(vp + v * 2);
            acc[dt] = wmma_bf16(pa, bv, acc[dt]);
        }
    }

    // ---- epilogue: normalize and store ctx [B,S,H*DK] bf16 ----
#pragma unroll
    for (int dt = 0; dt < 4; ++dt) {
#pragma unroll
        for (int v = 0; v < 8; ++v) {
            int r = v + (half ? 8 : 0);
            int q = qbase + r;
            int d = dt * 16 + l15;
            float val = acc[dt][v] / l_i[v];
            ctx[((size_t)(b * SS + q)) * DD + h * DK + d] = f32_to_bf16(val);
        }
    }
}

// ---------------- 6) output projection: ctx @ Wo -> f32 d_out ----------------
// Each wave computes a 16x64 tile: A-fragment reused across 4 WMMAs.
__global__ __launch_bounds__(256)
void gemm_out_kernel(const unsigned short* __restrict__ Ch,
                     const unsigned short* __restrict__ WoT,
                     float* __restrict__ out) {
    int lane = threadIdx.x & 31;
    int wave = threadIdx.x >> 5;
    int half = (lane >> 4) & 1;
    int l15  = lane & 15;
    int mtile = blockIdx.x;
    int nbase = (blockIdx.y * 8 + wave) * 64;   // 16 n-tile groups of 64

    const unsigned short* arow = Ch + (size_t)(mtile * 16 + l15) * DD;
    const unsigned short* brow[4];
#pragma unroll
    for (int dt = 0; dt < 4; ++dt)
        brow[dt] = WoT + (size_t)(nbase + dt * 16 + l15) * DD + (half ? 16 : 0);

    v8f acc[4] = {{}, {}, {}, {}};
    for (int kk = 0; kk < DD; kk += 32) {
        __builtin_prefetch((const void*)(arow + kk + 256), 0, 1);
        FragBF a;
#pragma unroll
        for (int v = 0; v < 8; ++v)
            a.u[v] = *(const unsigned int*)(arow + kk + koffA(v, half));
#pragma unroll
        for (int dt = 0; dt < 4; ++dt) {
            FragBF b;
#pragma unroll
            for (int v = 0; v < 8; ++v)
                b.u[v] = *(const unsigned int*)(brow[dt] + kk + v * 2);
            acc[dt] = wmma_bf16(a, b, acc[dt]);
        }
    }

#pragma unroll
    for (int dt = 0; dt < 4; ++dt) {
#pragma unroll
        for (int v = 0; v < 8; ++v) {
            int r = mtile * 16 + v + (half ? 8 : 0);
            out[(size_t)r * DD + nbase + dt * 16 + l15] = acc[dt][v];
        }
    }
}

// ---------------- host launcher ----------------
extern "C" void kernel_launch(void* const* d_in, const int* in_sizes, int n_in,
                              void* d_out, int out_size, void* d_ws, size_t ws_size,
                              hipStream_t stream) {
    const float* X     = (const float*)d_in[0];  // [B,S,D]
    const float* Wq    = (const float*)d_in[1];  // [D, H*DK]
    const float* Wk    = (const float*)d_in[2];
    const float* Wv    = (const float*)d_in[3];
    const float* Wo    = (const float*)d_in[4];  // [H*DK, D]
    const float* table = (const float*)d_in[5];  // [NBUK, H]

    float* out_attn = (float*)d_out;                          // B*S*D floats
    float* out_bias = (float*)d_out + (size_t)BB * SS * DD;   // H*S*S floats

    // workspace layout (bytes)
    unsigned char* ws = (unsigned char*)d_ws;
    unsigned short* Xh  = (unsigned short*)(ws);                    // 8 MiB
    unsigned short* WqT = (unsigned short*)(ws + (8u  << 20));      // 2 MiB each
    unsigned short* WkT = (unsigned short*)(ws + (10u << 20));
    unsigned short* WvT = (unsigned short*)(ws + (12u << 20));
    unsigned short* WoT = (unsigned short*)(ws + (14u << 20));
    unsigned short* Qb  = (unsigned short*)(ws + (16u << 20));      // 8 MiB
    unsigned short* Kb  = (unsigned short*)(ws + (24u << 20));      // 8 MiB
    unsigned short* Vt  = (unsigned short*)(ws + (32u << 20));      // 8 MiB
    unsigned short* Ctx = (unsigned short*)(ws + (40u << 20));      // 8 MiB

    const int nX = BB * SS * DD;

    // 1) converts / transposes
    conv_bf16_kernel<<<nX / 256, 256, 0, stream>>>(X, Xh, nX);
    transpose_bf16_kernel<<<(DD * DD) / 256, 256, 0, stream>>>(Wq, WqT);
    transpose_bf16_kernel<<<(DD * DD) / 256, 256, 0, stream>>>(Wk, WkT);
    transpose_bf16_kernel<<<(DD * DD) / 256, 256, 0, stream>>>(Wv, WvT);
    transpose_bf16_kernel<<<(DD * DD) / 256, 256, 0, stream>>>(Wo, WoT);

    // 2) position bias straight into d_out (also consumed by attention)
    bias_kernel<<<(SS * SS) / 256, 256, 0, stream>>>(table, out_bias);

    // 3) fused Q/K/V projections (X read once)
    dim3 qkvgrid(BB * SS / 16, DD / 128);
    fused_qkv_kernel<<<qkvgrid, 256, 0, stream>>>(Xh, WqT, WkT, WvT, Qb, Kb, Vt);

    // 4) flash attention
    dim3 agrid(SS / 64, HH, BB);
    attention_kernel<<<agrid, 128, 0, stream>>>(Qb, Kb, Vt, out_bias, Ctx);

    // 5) output projection (16x64 per-wave tiles)
    dim3 ogrid(BB * SS / 16, DD / 512);
    gemm_out_kernel<<<ogrid, 256, 0, stream>>>(Ctx, WoT, out_attn);
}
